// YOLODetectionLayer_16673063043233
// MI455X (gfx1250) — matrix operand order repaired and checked by reading, not compiled, not run on hardware
//
#include <hip/hip_runtime.h>
#include <hip/hip_bf16.h>

#define GRID_C    104
#define NB        9
#define NC        80
#define N_ANCH    (GRID_C*GRID_C*NB)      /* 97344 */
#define CH        (5+NC)                  /* 85 */
#define CANVAS    832
#define DET_T     0.02f
#define NMS_T     0.3f
#define MAX_BOXES 50
#define NSEL      (NC*MAX_BOXES)          /* 4000 = K dim, 125*32 */
#define KSTEPS    (NSEL/32)               /* 125 */
#define SUPP_WORDS ((N_ANCH+31)/32)       /* 3042 */
#define TILES     ((CANVAS/16)*(CANVAS/16)) /* 2704 */

typedef __attribute__((ext_vector_type(16))) _Float16 v16h;
typedef __attribute__((ext_vector_type(8)))  float    v8f;

// ---------------- decode: sigmoid/exp/softmax + threshold + argmax ----------------
__global__ void yolo_decode(const float* __restrict__ in, const float* __restrict__ anchors,
                            float4* __restrict__ fboxes, float* __restrict__ fscores,
                            int* __restrict__ fclasses) {
  int n = blockIdx.x * blockDim.x + threadIdx.x;
  if (n >= N_ANCH) return;
  int a    = n % NB;
  int cell = n / NB;
  int gx   = cell % GRID_C;
  int gy   = cell / GRID_C;
  const float* p = in + (long)n * CH;

  float bx = (1.0f / (1.0f + expf(-p[0])) + (float)gx) / (float)GRID_C;
  float by = (1.0f / (1.0f + expf(-p[1])) + (float)gy) / (float)GRID_C;
  float bw = expf(p[2]) * anchors[a * 2 + 0] / (float)GRID_C;
  float bh = expf(p[3]) * anchors[a * 2 + 1] / (float)GRID_C;
  fboxes[n] = make_float4(bx - 0.5f * bw, by - 0.5f * bh, bx + 0.5f * bw, by + 0.5f * bh);

  float obj = 1.0f / (1.0f + expf(-p[4]));
  float mx = -INFINITY;
  for (int c = 0; c < NC; ++c) mx = fmaxf(mx, p[5 + c]);
  float sum = 0.0f, best = -1.0f; int bc = 0;
  for (int c = 0; c < NC; ++c) {
    float e = expf(p[5 + c] - mx);
    sum += e;
    if (e > best) { best = e; bc = c; }   // strict > keeps first occurrence (jnp.argmax)
  }
  float score = obj * best / sum;
  if (score > DET_T) { fscores[n] = score; fclasses[n] = bc; }
  else               { fscores[n] = 0.0f;  fclasses[n] = 0;  }
}

// ---------------- per-class greedy NMS, suppression bitmask in LDS ----------------
__device__ __forceinline__ float iou_box(float4 a, float4 b) {
  float x1 = fmaxf(a.x, b.x), y1 = fmaxf(a.y, b.y);
  float x2 = fminf(a.z, b.z), y2 = fminf(a.w, b.w);
  float inter = fmaxf(x2 - x1, 0.0f) * fmaxf(y2 - y1, 0.0f);
  float a0 = (a.z - a.x) * (a.w - a.y);
  float a1 = (b.z - b.x) * (b.w - b.y);
  float un = a0 + a1 - inter;
  return un > 0.0f ? inter / un : 0.0f;
}

__device__ __forceinline__ void comb(float& bv, int& bi, float ov, int oi) {
  if (ov > bv || (ov == bv && (unsigned)oi < (unsigned)bi)) { bv = ov; bi = oi; }
}

__global__ __launch_bounds__(1024)
void yolo_nms(const float* __restrict__ fscores, const int* __restrict__ fclasses,
              const float4* __restrict__ fboxes,
              int* __restrict__ sel_idx, int* __restrict__ sel_valid) {
  const int c   = blockIdx.x;          // one class per workgroup
  const int tid = threadIdx.x;         // 1024 threads = 32 waves (wave32)
  __shared__ unsigned int supp[SUPP_WORDS];
  __shared__ float rbv[32];
  __shared__ int   rbi[32];
  __shared__ float s_bestv;
  __shared__ int   s_besti;

  for (int d = tid; d < SUPP_WORDS; d += 1024) supp[d] = 0u;
  __syncthreads();

  for (int it = 0; it < MAX_BOXES; ++it) {
    // ---- global argmax over live class-c candidates (first-index tie-break) ----
    float bv = 0.0f; int bi = -1;
    for (int i = tid; i < N_ANCH; i += 1024) {
      if (fclasses[i] == c) {
        float s = fscores[i];
        if (s > 0.0f && !((supp[i >> 5] >> (i & 31)) & 1u)) {
          if (s > bv) { bv = s; bi = i; }   // ascending i per thread -> first kept
        }
      }
    }
    for (int off = 16; off >= 1; off >>= 1) {
      float ov = __shfl_xor(bv, off, 32);
      int   oi = __shfl_xor(bi, off, 32);
      comb(bv, bi, ov, oi);
    }
    if ((tid & 31) == 0) { rbv[tid >> 5] = bv; rbi[tid >> 5] = bi; }
    __syncthreads();
    if (tid < 32) {
      float v = rbv[tid]; int i = rbi[tid];
      for (int off = 16; off >= 1; off >>= 1) {
        float ov = __shfl_xor(v, off, 32);
        int   oi = __shfl_xor(i, off, 32);
        comb(v, i, ov, oi);
      }
      if (tid == 0) { s_bestv = v; s_besti = i; }
    }
    __syncthreads();
    float bestv = s_bestv; int besti = s_besti;

    if (bestv <= 0.0f) {                 // all scores zero: remaining slots invalid
      if (tid == 0) {
        for (int j = it; j < MAX_BOXES; ++j) {
          sel_idx[c * MAX_BOXES + j] = 0; sel_valid[c * MAX_BOXES + j] = 0;
        }
      }
      break;                             // uniform (bestv is shared)
    }
    if (tid == 0) { sel_idx[c * MAX_BOXES + it] = besti; sel_valid[c * MAX_BOXES + it] = 1; }

    // ---- suppress: each thread owns whole bitmask dwords (no atomics) ----
    float4 pb = fboxes[besti];
    for (int d = tid; d < SUPP_WORDS; d += 1024) {
      unsigned int bits = 0u;
      int i0 = d << 5;
      int cnt = min(32, N_ANCH - i0);
      unsigned int cur = supp[d];
      for (int b = 0; b < cnt; ++b) {
        int i = i0 + b;
        if (((cur >> b) & 1u) == 0u && fclasses[i] == c && fscores[i] > 0.0f) {
          if (iou_box(pb, fboxes[i]) > NMS_T) bits |= (1u << b);
        }
      }
      if ((besti >> 5) == d) bits |= (1u << (besti & 31)); // explicit s[idx]=0
      supp[d] = cur | bits;
    }
    __syncthreads();
  }
}

// ---------------- selected boxes -> pixel coords (RNE round, clamp) ----------------
__device__ __forceinline__ int to_px(float v) {
  float t = rintf(v * (float)(CANVAS - 1));       // jnp.round = RNE
  t = fminf(fmaxf(t, 0.0f), (float)(CANVAS - 1));
  return (int)t;
}

__global__ void yolo_selpx(const int* __restrict__ sel_idx, const float4* __restrict__ fboxes,
                           int4* __restrict__ sel_px) {
  int k = blockIdx.x * blockDim.x + threadIdx.x;
  if (k >= NSEL) return;
  float4 b = fboxes[sel_idx[k]];
  // reference: r1=box[0], c1=box[1], r2=box[2], c2=box[3]
  sel_px[k] = make_int4(to_px(b.x), to_px(b.y), to_px(b.z), to_px(b.w));
}

// ---------------- build FP16 indicator matrices, K-contiguous [pixel][K] ----------------
// canvas[r][c] = sum_k RT[r][k]*CI[c][k] + RI[r][k]*CE[c][k]   (then >0)
__global__ void yolo_indicators(const int4* __restrict__ sel_px, const int* __restrict__ sel_valid,
                                _Float16* __restrict__ RT, _Float16* __restrict__ RI,
                                _Float16* __restrict__ CE, _Float16* __restrict__ CI) {
  long t = (long)blockIdx.x * blockDim.x + threadIdx.x;
  const long total = (long)CANVAS * NSEL;
  if (t >= total) return;
  int p = (int)(t / NSEL);
  int k = (int)(t % NSEL);
  int4 px = sel_px[k];
  int val = sel_valid[k];
  int rt = (p == px.x) + (p == px.z);                       // row-edge (0/1/2)
  int ce = (p == px.y) + (p == px.w);                       // col-edge (0/1/2)
  int ri = (val && p >= px.x && p <= px.z) ? 1 : 0;         // row-interval (carries val)
  int ci = (val && p >= px.y && p <= px.w) ? 1 : 0;         // col-interval (carries val)
  RT[t] = (_Float16)rt;
  RI[t] = (_Float16)ri;
  CE[t] = (_Float16)ce;
  CI[t] = (_Float16)ci;
}

// ---------------- draw canvas as dual GEMM via v_wmma_f32_16x16x32_f16 ----------------
// A (16x32 f16) layout: lane L holds row m0+(L&15); half h -> K = (h&7) + 16*(h>=8) + 8*(L>=16)
__device__ __forceinline__ v16h load_a(const _Float16* __restrict__ M, int row, int k0, int laneHi) {
  const uint4* p0 = (const uint4*)(M + (long)row * NSEL + k0 + laneHi * 8);
  const uint4* p1 = (const uint4*)(M + (long)row * NSEL + k0 + 16 + laneHi * 8);
  union { uint4 u[2]; v16h v; } u;
  u.u[0] = *p0; u.u[1] = *p1;
  return u.v;
}
// B (32x16 f16) layout: lane L holds col n0+(L&15); half h -> K = (h&7) + 16*(h>=8) + 8*(L>=16)
__device__ __forceinline__ v16h load_b(const _Float16* __restrict__ M, int col, int k0, int laneHi) {
  const uint4* p0 = (const uint4*)(M + (long)col * NSEL + k0 + laneHi * 8);
  const uint4* p1 = (const uint4*)(M + (long)col * NSEL + k0 + 16 + laneHi * 8);
  union { uint4 u[2]; v16h v; } u;
  u.u[0] = *p0; u.u[1] = *p1;
  return u.v;
}

__global__ __launch_bounds__(256)
void yolo_draw_wmma(const _Float16* __restrict__ RT, const _Float16* __restrict__ RI,
                    const _Float16* __restrict__ CE, const _Float16* __restrict__ CI,
                    float* __restrict__ out) {
  const int wave = threadIdx.x >> 5;            // 8 waves per block
  const int lane = threadIdx.x & 31;
  const int tile = blockIdx.x * 8 + wave;       // 338*8 = 2704 = 52*52 tiles exactly
  const int mt = tile / (CANVAS / 16);
  const int nt = tile % (CANVAS / 16);
  const int m0 = mt * 16, n0 = nt * 16;
  const int laneHi = lane >> 4;
  const int rowA = m0 + (lane & 15);
  const int colB = n0 + (lane & 15);

  v8f acc = {};
  for (int kt = 0; kt < KSTEPS; ++kt) {
    const int k0 = kt * 32;
    v16h a_rt = load_a(RT, rowA, k0, laneHi);
    v16h a_ri = load_a(RI, rowA, k0, laneHi);
    v16h b_ci = load_b(CI, colB, k0, laneHi);
    v16h b_ce = load_b(CE, colB, k0, laneHi);
    acc = __builtin_amdgcn_wmma_f32_16x16x32_f16(false, a_rt, false, b_ci,
                                                 (short)0, acc, false, false);
    acc = __builtin_amdgcn_wmma_f32_16x16x32_f16(false, a_ri, false, b_ce,
                                                 (short)0, acc, false, false);
  }
  // C/D layout: VGPR v, lane L -> M = v + 8*(L>=16), N = L&15 ; binarize on store
  const int col = n0 + (lane & 15);
#pragma unroll
  for (int v = 0; v < 8; ++v) {
    int row = m0 + v + (laneHi ? 8 : 0);
    out[(long)row * CANVAS + col] = (acc[v] > 0.0f) ? 1.0f : 0.0f;
  }
}

// ---------------- host-side launcher ----------------
static inline char* bump(char*& p, size_t bytes) {
  char* r = p;
  p += (bytes + 255) & ~(size_t)255;
  return r;
}

extern "C" void kernel_launch(void* const* d_in, const int* in_sizes, int n_in,
                              void* d_out, int out_size, void* d_ws, size_t ws_size,
                              hipStream_t stream) {
  const float* inputs  = (const float*)d_in[0];   // (1,104,104,9,85) f32
  const float* anchors = (const float*)d_in[1];   // (9,2) f32
  float* out = (float*)d_out;                     // (1,832,832,1) f32

  char* w = (char*)d_ws;
  float4*   fboxes   = (float4*)  bump(w, sizeof(float4) * N_ANCH);
  float*    fscores  = (float*)   bump(w, sizeof(float)  * N_ANCH);
  int*      fclasses = (int*)     bump(w, sizeof(int)    * N_ANCH);
  int*      sel_idx  = (int*)     bump(w, sizeof(int)    * NSEL);
  int*      sel_val  = (int*)     bump(w, sizeof(int)    * NSEL);
  int4*     sel_px   = (int4*)    bump(w, sizeof(int4)   * NSEL);
  _Float16* RT = (_Float16*)      bump(w, sizeof(_Float16) * (size_t)CANVAS * NSEL);
  _Float16* RI = (_Float16*)      bump(w, sizeof(_Float16) * (size_t)CANVAS * NSEL);
  _Float16* CE = (_Float16*)      bump(w, sizeof(_Float16) * (size_t)CANVAS * NSEL);
  _Float16* CI = (_Float16*)      bump(w, sizeof(_Float16) * (size_t)CANVAS * NSEL);

  yolo_decode<<<(N_ANCH + 255) / 256, 256, 0, stream>>>(inputs, anchors, fboxes, fscores, fclasses);
  yolo_nms<<<NC, 1024, 0, stream>>>(fscores, fclasses, fboxes, sel_idx, sel_val);
  yolo_selpx<<<(NSEL + 255) / 256, 256, 0, stream>>>(sel_idx, fboxes, sel_px);
  {
    long total = (long)CANVAS * NSEL;
    int blocks = (int)((total + 255) / 256);
    yolo_indicators<<<blocks, 256, 0, stream>>>(sel_px, sel_val, RT, RI, CE, CI);
  }
  yolo_draw_wmma<<<TILES / 8, 256, 0, stream>>>(RT, RI, CE, CI, out);
}
